// SSAMultiHeadAttention__75634374082828
// MI455X (gfx1250) — 16-bit and 8-bit matrix operand order repaired and checked by reading, compiled, not run on hardware
//
#include <hip/hip_runtime.h>

#define Bq 4
#define Nn 1024
#define Dd 1024
#define Hh 16
#define DHd 64
#define BN  4096

typedef float  v8f  __attribute__((ext_vector_type(8)));
typedef int    v8i  __attribute__((ext_vector_type(8)));
typedef __bf16 v16bf __attribute__((ext_vector_type(16)));
typedef unsigned int u32;
typedef u32 u32x2 __attribute__((ext_vector_type(2)));
typedef u32 u32x4 __attribute__((ext_vector_type(4)));

__device__ __forceinline__ unsigned short f2bf(float f) {
    u32 x = __float_as_uint(f);
    u32 r = x + 0x7FFFu + ((x >> 16) & 1u);
    return (unsigned short)(r >> 16);
}

union BF16Frag { v16bf v; u32x4 q[2]; unsigned short u[16]; };
union Frag8    { v8i v; u32x4 q[2]; u32x2 d[4]; };

__device__ __forceinline__ v8i load_a8_packed(const unsigned char* lane_base) {
    Frag8 u;
    u.q[0] = *(const u32x4*)(lane_base);
    u.q[1] = *(const u32x4*)(lane_base + 16);
    return u.v;
}

__device__ __forceinline__ v8i load_b8(const unsigned char* base, int hi) {
    Frag8 u;
    u.d[0] = *(const u32x2*)(base + hi * 8);
    u.d[1] = *(const u32x2*)(base + 16 + hi * 8);
    u.d[2] = *(const u32x2*)(base + 32 + hi * 8);
    u.d[3] = *(const u32x2*)(base + 48 + hi * 8);
    return u.v;
}

__global__ __launch_bounds__(256) void pack_prep(
    const float* __restrict__ Wq, const float* __restrict__ Wk,
    const float* __restrict__ Wv, const float* __restrict__ Wo,
    const float* __restrict__ mask,
    unsigned short* __restrict__ WpBase, unsigned char* __restrict__ mask8)
{
    int i = blockIdx.x * 256 + threadIdx.x;
    int which = blockIdx.y;
    if (which == 4) {
        mask8[i] = (mask[i] != 0.0f) ? 0xFF : 0;
        return;
    }
    const float* W = (which == 0) ? Wq : (which == 1) ? Wk : (which == 2) ? Wv : Wo;
    unsigned short* Wp = WpBase + (size_t)which * (Dd * Dd);
    int h    = i & 15;
    int lane = (i >> 4) & 31;
    int nt   = (i >> 9) & 63;
    int kt   = i >> 15;
    int k = kt * 32 + (lane >> 4) * 16 + h;
    int n = nt * 16 + (lane & 15);
    Wp[i] = f2bf(W[k * Dd + n]);
}

template <int MODE>
__global__ __launch_bounds__(256) void gemm_ln_lif(
    const float* __restrict__ Af, const unsigned char* __restrict__ A8,
    const unsigned short* __restrict__ Wp, const float* __restrict__ bias,
    const float* __restrict__ gamma, const float* __restrict__ beta,
    unsigned char* __restrict__ out8, float* __restrict__ outf)
{
    union SM {
        struct { unsigned short a[16][32]; } g;
        struct {
            float rs[16][128]; float rq[16][128];
            float ps[16][16];  float pq[16][16];
            float mu[16];      float rstd[16];
        } l;
    };
    __shared__ SM sm;

    const int tid  = threadIdx.x;
    const int wave = tid >> 5;
    const int lane = tid & 31;
    const int hi   = lane >> 4;
    const int l15  = lane & 15;
    const int rowBase = blockIdx.x * 16;

    v8f acc[8] = {};

    for (int k0 = 0; k0 < Dd; k0 += 32) {
        for (int i = tid; i < 16 * 32; i += 256) {
            int r = i >> 5, c = i & 31;
            int idx = (rowBase + r) * Dd + k0 + c;
            if (MODE == 3)
                sm.g.a[r][c] = A8[idx] ? (unsigned short)0x3F80 : (unsigned short)0;
            else
                sm.g.a[r][c] = f2bf(Af[idx]);
        }
        __syncthreads();

        BF16Frag af;
        af.q[0] = *(const u32x4*)&sm.g.a[l15][hi * 16];
        af.q[1] = *(const u32x4*)&sm.g.a[l15][hi * 16 + 8];

        const int kt = k0 >> 5;
        #pragma unroll
        for (int t = 0; t < 8; ++t) {
            const unsigned short* wp =
                Wp + (((size_t)(kt * 64 + wave * 8 + t) * 32 + lane) << 4);
            BF16Frag bf;
            bf.q[0] = *(const u32x4*)(wp);
            bf.q[1] = *(const u32x4*)(wp + 8);
            acc[t] = __builtin_amdgcn_wmma_f32_16x16x32_bf16(
                false, af.v, false, bf.v, (short)0, acc[t], false, false);
        }
        __syncthreads();
    }

    #pragma unroll
    for (int i = 0; i < 8; ++i) {
        int r = hi * 8 + i;
        float s = 0.f, q = 0.f;
        #pragma unroll
        for (int t = 0; t < 8; ++t) {
            int c = wave * 128 + t * 16 + l15;
            float y = acc[t][i] + bias[c];
            s += y; q += y * y;
        }
        sm.l.rs[r][wave * 16 + l15] = s;
        sm.l.rq[r][wave * 16 + l15] = q;
    }
    __syncthreads();
    {
        int rr = tid >> 4, j = tid & 15;
        float s = 0.f, q = 0.f;
        #pragma unroll
        for (int e = 0; e < 8; ++e) { s += sm.l.rs[rr][j * 8 + e]; q += sm.l.rq[rr][j * 8 + e]; }
        sm.l.ps[rr][j] = s; sm.l.pq[rr][j] = q;
    }
    __syncthreads();
    if ((tid & 15) == 0) {
        int rr = tid >> 4;
        float s = 0.f, q = 0.f;
        #pragma unroll
        for (int e = 0; e < 16; ++e) { s += sm.l.ps[rr][e]; q += sm.l.pq[rr][e]; }
        float m = s * (1.0f / 1024.0f);
        float v = q * (1.0f / 1024.0f) - m * m;
        sm.l.mu[rr]   = m;
        sm.l.rstd[rr] = rsqrtf(v + 1e-5f);
    }
    __syncthreads();

    #pragma unroll
    for (int t = 0; t < 8; ++t) {
        int c = wave * 128 + t * 16 + l15;
        float g = gamma[c], be = beta[c], bi = bias[c];
        #pragma unroll
        for (int i = 0; i < 8; ++i) {
            int r = hi * 8 + i;
            float y = (acc[t][i] + bi - sm.l.mu[r]) * sm.l.rstd[r] * g + be;
            int spike = (y >= 1.0f) ? 1 : 0;
            int row = rowBase + r;
            if (MODE == 0) {
                out8[(size_t)row * Dd + c] = (unsigned char)spike;
            } else if (MODE == 1) {
                int bb = row >> 10, n = row & 1023;
                int hh = c >> 6,   dh = c & 63;
                int r2  = dh >> 4;
                int hi2 = (dh >> 3) & 1;
                int off = dh & 7;
                size_t idx = ((((size_t)(bb * Hh + hh) * 64 + (n >> 4)) * 32)
                              + hi2 * 16 + (n & 15)) * 32 + r2 * 8 + off;
                out8[idx] = (unsigned char)spike;
            } else if (MODE == 2) {
                int bb = row >> 10, n = row & 1023;
                int hh = c >> 6,   dh = c & 63;
                out8[((size_t)((bb * Hh + hh) * DHd + dh) << 10) + n] = (unsigned char)spike;
            } else {
                outf[(size_t)row * Dd + c] = spike ? 1.0f : 0.0f;
            }
        }
    }
}

__global__ __launch_bounds__(32) void spike_attn(
    const unsigned char* __restrict__ Qs, const unsigned char* __restrict__ Kp,
    const unsigned char* __restrict__ VsT, const unsigned char* __restrict__ mask8,
    unsigned char* __restrict__ Cs)
{
    __shared__ __align__(16) unsigned char Sp[16 * 1024];

    const int lane = threadIdx.x;
    const int hi   = lane >> 4;
    const int l15  = lane & 15;
    const int nt0  = blockIdx.x;
    const int h    = blockIdx.y;
    const int b    = blockIdx.z;

    const unsigned char* qbase = Qs + ((size_t)(b * Nn + nt0 * 16 + l15)) * Dd + h * DHd;
    v8i bq = load_b8(qbase, hi);

    const unsigned char* kpBase = Kp + ((size_t)(b * Hh + h) << 16);
    const unsigned char* mrow = mask8 + (size_t)(nt0 * 16 + l15) * Nn;

    for (int nt = 0; nt < Nn / 16; ++nt) {
        v8i ak = load_a8_packed(kpBase + (size_t)nt * 1024 + lane * 32);
        v8i c = {};
        c = __builtin_amdgcn_wmma_i32_16x16x64_iu8(false, ak, false, bq, c, false, false);

        u32x2 mw = *(const u32x2*)(mrow + nt * 16 + hi * 8);
        u32 p0 = (u32)c[0] | ((u32)c[1] << 8) | ((u32)c[2] << 16) | ((u32)c[3] << 24);
        u32 p1 = (u32)c[4] | ((u32)c[5] << 8) | ((u32)c[6] << 16) | ((u32)c[7] << 24);
        u32x2 st; st.x = p0 & mw.x; st.y = p1 & mw.y;

        *(u32x2*)&Sp[((nt >> 2) << 10) + lane * 32 + (nt & 3) * 8] = st;
    }
    __syncthreads();

    #pragma unroll
    for (int dt = 0; dt < 4; ++dt) {
        v8i acc = {};
        const unsigned char* vb0 = VsT + ((size_t)((b * Hh + h) * DHd + dt * 16 + l15)) * Nn;
        for (int kc = 0; kc < 16; ++kc) {
            v8i as = load_a8_packed(&Sp[(kc << 10) + lane * 32]);
            v8i bv = load_b8(vb0 + kc * 64, hi);
            acc = __builtin_amdgcn_wmma_i32_16x16x64_iu8(false, as, false, bv, acc, false, false);
        }
        #pragma unroll
        for (int i = 0; i < 8; ++i) {
            int m  = hi * 8 + i;
            int dh = dt * 16 + l15;
            unsigned char sp = (acc[i] >= 4) ? 1 : 0;
            Cs[((size_t)(b * Nn + nt0 * 16 + m)) * Dd + h * DHd + dh] = sp;
        }
    }
}

extern "C" void kernel_launch(void* const* d_in, const int* in_sizes, int n_in,
                              void* d_out, int out_size, void* d_ws, size_t ws_size,
                              hipStream_t stream) {
    const float* query = (const float*)d_in[0];
    const float* amask = (const float*)d_in[1];
    const float* Wq = (const float*)d_in[2];  const float* bq = (const float*)d_in[3];
    const float* Wk = (const float*)d_in[4];  const float* bk = (const float*)d_in[5];
    const float* Wv = (const float*)d_in[6];  const float* bv = (const float*)d_in[7];
    const float* Wo = (const float*)d_in[8];  const float* bo = (const float*)d_in[9];
    const float* lnq_g = (const float*)d_in[10]; const float* lnq_b = (const float*)d_in[11];
    const float* lnk_g = (const float*)d_in[12]; const float* lnk_b = (const float*)d_in[13];
    const float* lnv_g = (const float*)d_in[14]; const float* lnv_b = (const float*)d_in[15];
    const float* lno_g = (const float*)d_in[16]; const float* lno_b = (const float*)d_in[17];

    unsigned char* ws = (unsigned char*)d_ws;
    const size_t SEG = (size_t)BN * Dd;
    unsigned char* Qs  = ws;
    unsigned char* Kp  = ws + SEG;
    unsigned char* VsT = ws + 2 * SEG;
    unsigned char* Cs  = ws + 3 * SEG;
    unsigned short* Wp = (unsigned short*)(ws + 4 * SEG);
    unsigned char* mask8 = ws + 4 * SEG + 4 * SEG * sizeof(unsigned short);

    pack_prep<<<dim3((Dd * Dd) / 256, 5), 256, 0, stream>>>(
        Wq, Wk, Wv, Wo, amask, Wp, mask8);

    dim3 gGemm(BN / 16);
    gemm_ln_lif<0><<<gGemm, 256, 0, stream>>>(query, nullptr, Wp + 0 * (size_t)Dd * Dd,
                                              bq, lnq_g, lnq_b, Qs,  nullptr);
    gemm_ln_lif<1><<<gGemm, 256, 0, stream>>>(query, nullptr, Wp + 1 * (size_t)Dd * Dd,
                                              bk, lnk_g, lnk_b, Kp,  nullptr);
    gemm_ln_lif<2><<<gGemm, 256, 0, stream>>>(query, nullptr, Wp + 2 * (size_t)Dd * Dd,
                                              bv, lnv_g, lnv_b, VsT, nullptr);

    spike_attn<<<dim3(Nn / 16, Hh, Bq), 32, 0, stream>>>(Qs, Kp, VsT, mask8, Cs);

    gemm_ln_lif<3><<<gGemm, 256, 0, stream>>>(nullptr, Cs, Wp + 3 * (size_t)Dd * Dd,
                                              bo, lno_g, lno_b, nullptr, (float*)d_out);
}
